// Attention_1228360646767
// MI455X (gfx1250) — matrix operand order repaired and checked by reading, not compiled, not run on hardware
//
#include <hip/hip_runtime.h>
#include <hip/hip_bf16.h>

// ---------------------------------------------------------------------------
// MI455X (gfx1250) multi-head attention, bf16 WMMA pipeline + TDM staging.
// B=2, QL=512, KL=8192, D=1024, H=8, HD=128.
// ---------------------------------------------------------------------------

typedef __bf16 bf16_t;
typedef __attribute__((ext_vector_type(16))) __bf16 v16bf;
typedef __attribute__((ext_vector_type(8)))  __bf16 v8bf;
typedef __attribute__((ext_vector_type(8)))  float  v8f;
typedef __attribute__((ext_vector_type(4)))  unsigned int v4u;
typedef __attribute__((ext_vector_type(8)))  int v8i;
typedef __attribute__((ext_vector_type(4)))  int v4i;

#define B_   2
#define QL_  512
#define KL_  8192
#define D_   1024
#define H_   8
#define HD_  128
#define NSEG 8                      // split-K segments over KL
#define SEGC (KL_ / NSEG / 64)      // 64-key chunks per segment = 16

__device__ __forceinline__ v8f vzero8() {
  v8f v;
#pragma unroll
  for (int i = 0; i < 8; ++i) v[i] = 0.0f;
  return v;
}

__device__ __forceinline__ v16bf cat16(v8bf lo, v8bf hi) {
  v16bf r;
#pragma unroll
  for (int i = 0; i < 8; ++i) { r[i] = lo[i]; r[i + 8] = hi[i]; }
  return r;
}

// 16x32 bf16 A-fragment from a row-major slab (stride in halfs).
// lanes 0-15 -> row M=lane, K[0..7]|K[16..23]; lanes 16-31 -> K[8..15]|K[24..31].
__device__ __forceinline__ v16bf load_frag_a(const bf16_t* base, int stride, int lane) {
  const bf16_t* p = base + (lane & 15) * stride + ((lane >> 4) << 3);
  return cat16(*(const v8bf*)(p), *(const v8bf*)(p + 16));
}

// 32x16 bf16 B-fragment. Source slab row n = B column n (row-major N x K).
// lanes 0-15 -> col N=lane, K[0..7]|K[16..23]; lanes 16-31 -> col N=lane-16, K[8..15]|K[24..31].
__device__ __forceinline__ v16bf load_frag_b(const bf16_t* base, int stride, int lane) {
  const bf16_t* p = base + (lane & 15) * stride + ((lane >> 4) << 3);
  return cat16(*(const v8bf*)(p), *(const v8bf*)(p + 16));
}

__device__ __forceinline__ v8f wmma_bf16(v16bf a, v16bf b, v8f c) {
  return __builtin_amdgcn_wmma_f32_16x16x32_bf16(false, a, false, b, (short)0, c,
                                                 false, false);
}

// ---------------------------------------------------------------------------
// fp32 -> bf16 conversion, 8 elements per thread (b128 load / b128 store)
// ---------------------------------------------------------------------------
__global__ __launch_bounds__(256) void cvt_f32_bf16_kernel(
    const float* __restrict__ s, bf16_t* __restrict__ d, int n8) {
  int i = blockIdx.x * 256 + threadIdx.x;
  if (i >= n8) return;
  const float* p = s + (size_t)i * 8;
  v8bf o;
#pragma unroll
  for (int j = 0; j < 8; ++j) o[j] = (bf16_t)p[j];
  *(v8bf*)(d + (size_t)i * 8) = o;
}

// ---------------------------------------------------------------------------
// Pack bool mask bytes into one u64 bitword per 64 keys (vector reads).
// ---------------------------------------------------------------------------
__global__ __launch_bounds__(256) void pack_mask_kernel(
    const unsigned char* __restrict__ m, unsigned long long* __restrict__ out,
    int n64) {
  int i = blockIdx.x * 256 + threadIdx.x;
  if (i >= n64) return;
  const unsigned long long* p = (const unsigned long long*)(m + (size_t)i * 64);
  unsigned long long w = 0ull;
#pragma unroll
  for (int g = 0; g < 8; ++g) {
    unsigned long long v = p[g];
#pragma unroll
    for (int j = 0; j < 8; ++j)
      w |= (unsigned long long)(((v >> (8 * j)) & 0xFFull) != 0) << (g * 8 + j);
  }
  out[i] = w;
}

// ---------------------------------------------------------------------------
// C[M,N] = A[M,K] * W[N,K]^T + bias[N].  128x128 block, 8 wave32s, each wave
// a 32x64 tile (2x4 WMMA tiles).  Ping-pong LDS (TK=32, 2x20KB): one barrier
// per K-step; next tile's global loads issue before the WMMA block so global
// latency overlaps compute.
// ---------------------------------------------------------------------------
template <bool OUT_F32>
__global__ __launch_bounds__(256) void gemm_bias_kernel(
    const bf16_t* __restrict__ A, const bf16_t* __restrict__ W,
    const float* __restrict__ bias, void* __restrict__ Cout,
    int M, int N, int K) {
  constexpr int TK  = 32;
  constexpr int LDT = 40;  // 32 + 8 halfs padding (80B rows, conflict-free)
  __shared__ bf16_t As[2][128 * LDT];
  __shared__ bf16_t Bs[2][128 * LDT];

  const int tid  = threadIdx.x;
  const int lane = tid & 31;
  const int wave = tid >> 5;
  const int wm   = wave >> 1;
  const int wn   = wave & 1;
  const int m0   = blockIdx.y * 128;
  const int n0   = blockIdx.x * 128;

  const int srow = tid >> 2;        // 0..63 (2 passes of 64 rows)
  const int scol = (tid & 3) * 8;   // 0,8,16,24

  v8f acc[2][4];
#pragma unroll
  for (int i = 0; i < 2; ++i)
#pragma unroll
    for (int j = 0; j < 4; ++j) acc[i][j] = vzero8();

  // Prologue: stage tile 0 into buffer 0.
  v8bf ra[2], rb[2];
#pragma unroll
  for (int p = 0; p < 2; ++p) {
    int r = srow + p * 64;
    ra[p] = *(const v8bf*)(A + (size_t)(m0 + r) * K + scol);
    rb[p] = *(const v8bf*)(W + (size_t)(n0 + r) * K + scol);
  }
#pragma unroll
  for (int p = 0; p < 2; ++p) {
    int r = srow + p * 64;
    *(v8bf*)&As[0][r * LDT + scol] = ra[p];
    *(v8bf*)&Bs[0][r * LDT + scol] = rb[p];
  }

  int cur = 0;
  const int nk = K / TK;
  for (int kk = 0; kk < nk; ++kk) {
    __syncthreads();  // buf[cur] ready; reads of buf[cur^1] (iter kk-1) done

    if (kk + 1 < nk) {  // issue next tile's global loads early
      int k0 = (kk + 1) * TK;
#pragma unroll
      for (int p = 0; p < 2; ++p) {
        int r = srow + p * 64;
        ra[p] = *(const v8bf*)(A + (size_t)(m0 + r) * K + k0 + scol);
        rb[p] = *(const v8bf*)(W + (size_t)(n0 + r) * K + k0 + scol);
      }
    }

    v16bf af[2];
#pragma unroll
    for (int i = 0; i < 2; ++i)
      af[i] = load_frag_a(&As[cur][(wm * 32 + i * 16) * LDT], LDT, lane);
#pragma unroll
    for (int j = 0; j < 4; ++j) {
      v16bf bfrag = load_frag_b(&Bs[cur][(wn * 64 + j * 16) * LDT], LDT, lane);
#pragma unroll
      for (int i = 0; i < 2; ++i) acc[i][j] = wmma_bf16(af[i], bfrag, acc[i][j]);
    }

    if (kk + 1 < nk) {
#pragma unroll
      for (int p = 0; p < 2; ++p) {
        int r = srow + p * 64;
        *(v8bf*)&As[cur ^ 1][r * LDT + scol] = ra[p];
        *(v8bf*)&Bs[cur ^ 1][r * LDT + scol] = rb[p];
      }
      cur ^= 1;
    }
  }

  const int nl = lane & 15;
  const int mh = (lane >> 4) * 8;
#pragma unroll
  for (int j = 0; j < 4; ++j) {
    int n = n0 + wn * 64 + j * 16 + nl;
    float bj = bias ? bias[n] : 0.0f;
#pragma unroll
    for (int i = 0; i < 2; ++i) {
#pragma unroll
      for (int r = 0; r < 8; ++r) {
        int m = m0 + wm * 32 + i * 16 + mh + r;
        float v = acc[i][j][r] + bj;
        if (OUT_F32) ((float*)Cout)[(size_t)m * N + n] = v;
        else         ((bf16_t*)Cout)[(size_t)m * N + n] = (bf16_t)v;
      }
    }
  }
}

// ---------------------------------------------------------------------------
// Split-K flash attention segment. Block = (seg, b, h, 128 q rows), 8 waves;
// wave owns 16 q rows, iterates SEGC chunks of 64 keys.
// K chunk staged by the Tensor Data Mover; V staged manually (transposed).
// Emits unnormalized partial O + per-row (m, l) stats.
// ---------------------------------------------------------------------------
#define KC  64
#define LDK 136   // Ks row: 128 d + 8 pad halfs (TDM pad: 4 dwords per 64)
#define LDV 72    // Vst / Ps row: 64 keys + 8 pad halfs

__global__ __launch_bounds__(256) void flash_seg_kernel(
    const bf16_t* __restrict__ Q, const bf16_t* __restrict__ Kp,
    const bf16_t* __restrict__ Vp, const unsigned long long* __restrict__ mw,
    float* __restrict__ Opart, float* __restrict__ mpart,
    float* __restrict__ lpart) {
  __shared__ bf16_t Ks[KC * LDK];       // [key][d], filled by TDM
  __shared__ bf16_t Vst[HD_ * LDV];     // [d][key]
  __shared__ bf16_t Ps[8][16 * LDV];    // per-wave P tile [qrow][key]

  const int tid  = threadIdx.x;
  const int lane = tid & 31;
  const int wave = tid >> 5;

  int blk = blockIdx.x;
  const int qt  = blk & 3;
  const int h   = (blk >> 2) & 7;
  const int b   = (blk >> 5) & 1;
  const int seg = blk >> 6;

  const int qbase = qt * 128 + wave * 16;
  const int nl = lane & 15;
  const int mh = (lane >> 4) * 8;

  // Q fragments (16 rows x 128 d = 4 k-steps) straight from global.
  const bf16_t* qptr = Q + ((size_t)(b * QL_ + qbase)) * D_ + h * HD_;
  v16bf qf[4];
#pragma unroll
  for (int ks = 0; ks < 4; ++ks) {
    const bf16_t* p = qptr + (size_t)nl * D_ + ks * 32 + ((lane >> 4) << 3);
    qf[ks] = cat16(*(const v8bf*)p, *(const v8bf*)(p + 16));
  }

  v8f Ot[8];
#pragma unroll
  for (int t = 0; t < 8; ++t) Ot[t] = vzero8();
  float mrow[8], lrow[8];
  int anyv[8];
#pragma unroll
  for (int r = 0; r < 8; ++r) { mrow[r] = -__builtin_inff(); lrow[r] = 0.0f; anyv[r] = 0; }

  const float scale = 0.088388347648318447f;  // 1/sqrt(128)

  const int skey = tid >> 4;
  const int sseg = tid & 15;
  const bf16_t* kbase_g = Kp + ((size_t)b * KL_) * D_ + h * HD_;
  const bf16_t* vbase_g = Vp + ((size_t)b * KL_) * D_ + h * HD_;

  for (int c = 0; c < SEGC; ++c) {
    const int cg = seg * SEGC + c;  // global 64-key chunk index
    __syncthreads();

    // --- TDM: K chunk (64 keys x 128 d bf16, tensor row stride D=1024) ---
    if (wave == 0) {
      unsigned long long ga =
          (unsigned long long)(const void*)(kbase_g + (size_t)cg * KC * D_);
      v4u g0;
      g0[0] = 1u;                                    // count=1 (valid D#)
      g0[1] = (unsigned int)(size_t)(void*)&Ks[0];   // LDS byte offset
      g0[2] = (unsigned int)(ga & 0xFFFFFFFFu);
      g0[3] = (unsigned int)((ga >> 32) & 0x01FFFFFFu) | (2u << 30);  // type=2
      v8i g1;
      // data_size=2B (1<<16), pad_enable (1<<20), pad_interval=5 -> 64 dwords,
      // pad_amount=3 -> 4 dwords => LDS row stride 136 halfs.
      g1[0] = (1 << 16) | (1 << 20) | (5 << 22) | (3 << 25);
      g1[1] = (HD_ << 16);   // tensor_dim0[15:0] in bits 63:48
      g1[2] = (KC << 16);    // tensor_dim1[15:0] in bits 95:80
      g1[3] = (HD_ << 16);   // tile_dim0=128 in bits 127:112
      g1[4] = KC;            // tile_dim1=64, tile_dim2=0
      g1[5] = D_;            // tensor_dim0_stride = 1024 elements
      g1[6] = 0;
      g1[7] = 0;
      v4i gz = {0, 0, 0, 0};
#if defined(__clang_major__) && (__clang_major__ >= 23)
      v8i gz8 = {0, 0, 0, 0, 0, 0, 0, 0};
      __builtin_amdgcn_tensor_load_to_lds(g0, g1, gz, gz, gz8, 0);
#else
      __builtin_amdgcn_tensor_load_to_lds(g0, g1, gz, gz, 0);
#endif
    }

    // --- V chunk staged transposed: Vst[d][key] (overlaps the TDM) ---
#pragma unroll
    for (int p = 0; p < 4; ++p) {
      int key = p * 16 + skey;
      v8bf vv = *(const v8bf*)(vbase_g + (size_t)(cg * KC + key) * D_ + sseg * 8);
#pragma unroll
      for (int e = 0; e < 8; ++e) Vst[(sseg * 8 + e) * LDV + key] = vv[e];
    }
    if (c + 1 < SEGC) {  // global_prefetch_b8 next chunk
      __builtin_prefetch(vbase_g + (size_t)((cg + 1) * KC + skey) * D_ + sseg * 8, 0, 1);
      __builtin_prefetch(kbase_g + (size_t)((cg + 1) * KC + skey) * D_ + sseg * 8, 0, 1);
    }
    if (wave == 0) __builtin_amdgcn_s_wait_tensorcnt(0);
    __syncthreads();

    // S(16x64) = Q(16x128) * Kchunk^T
    v8f St[4];
#pragma unroll
    for (int t = 0; t < 4; ++t) St[t] = vzero8();
#pragma unroll
    for (int t = 0; t < 4; ++t)
#pragma unroll
      for (int ks = 0; ks < 4; ++ks) {
        v16bf kfrag = load_frag_b(&Ks[(t * 16) * LDK + ks * 32], LDK, lane);
        St[t] = wmma_bf16(qf[ks], kfrag, St[t]);
      }

    // Online softmax; mask is one u64 bitword per row per chunk.
    float alpha[8];
#pragma unroll
    for (int r = 0; r < 8; ++r) {
      const int m = mh + r;
      unsigned long long w =
          mw[(size_t)(b * QL_ + qbase + m) * (KL_ / 64) + cg];
      anyv[r] |= (w != 0ull);
      float sv[4], smax = -__builtin_inff();
#pragma unroll
      for (int t = 0; t < 4; ++t) {
        int ok = (int)((w >> (nl + t * 16)) & 1ull);
        float s = ok ? St[t][r] * scale : -1.0e30f;
        sv[t] = s;
        smax = fmaxf(smax, s);
      }
#pragma unroll
      for (int off = 1; off < 16; off <<= 1)
        smax = fmaxf(smax, __shfl_xor(smax, off, 32));
      float mn = fmaxf(mrow[r], smax);
      alpha[r] = __expf(mrow[r] - mn);
      float lsum = 0.0f;
#pragma unroll
      for (int t = 0; t < 4; ++t) {
        float pv = __expf(sv[t] - mn);
        lsum += pv;
        Ps[wave][m * LDV + t * 16 + nl] = (bf16_t)pv;
      }
#pragma unroll
      for (int off = 1; off < 16; off <<= 1)
        lsum += __shfl_xor(lsum, off, 32);
      lrow[r] = lrow[r] * alpha[r] + lsum;
      mrow[r] = mn;
    }

    // O = O*alpha + P(16x64) * V(64x128)
#pragma unroll
    for (int t = 0; t < 8; ++t)
#pragma unroll
      for (int r = 0; r < 8; ++r) Ot[t][r] *= alpha[r];

    v16bf af[2];
#pragma unroll
    for (int ks = 0; ks < 2; ++ks)
      af[ks] = load_frag_a(&Ps[wave][ks * 32], LDV, lane);
#pragma unroll
    for (int t = 0; t < 8; ++t)
#pragma unroll
      for (int ks = 0; ks < 2; ++ks) {
        v16bf vfrag = load_frag_b(&Vst[(t * 16) * LDV + ks * 32], LDV, lane);
        Ot[t] = wmma_bf16(af[ks], vfrag, Ot[t]);
      }
  }

  // Emit partials (unnormalized) + stats. anyv is uniform per row already.
  float* opt = Opart + (size_t)seg * (B_ * QL_ * D_) +
               ((size_t)(b * QL_ + qbase)) * D_ + h * HD_;
#pragma unroll
  for (int t = 0; t < 8; ++t) {
    int d = t * 16 + nl;
#pragma unroll
    for (int r = 0; r < 8; ++r) {
      int m = mh + r;
      opt[(size_t)m * D_ + d] = anyv[r] ? Ot[t][r] : 0.0f;
    }
  }
  if (nl == 0) {
#pragma unroll
    for (int r = 0; r < 8; ++r) {
      int m = mh + r;
      size_t si = (((size_t)seg * B_ + b) * H_ + h) * QL_ + (qbase + m);
      mpart[si] = anyv[r] ? mrow[r] : -1.0e38f;
      lpart[si] = anyv[r] ? lrow[r] : 0.0f;
    }
  }
}

// ---------------------------------------------------------------------------
// Merge split-K partials: out = sum_s exp(m_s - m*) O_s / sum_s exp(m_s-m*) l_s
// ---------------------------------------------------------------------------
__global__ __launch_bounds__(256) void combine_kernel(
    const float* __restrict__ Opart, const float* __restrict__ mpart,
    const float* __restrict__ lpart, bf16_t* __restrict__ ao) {
  size_t idx = (size_t)blockIdx.x * 256 + threadIdx.x;  // over B*QL*D
  if (idx >= (size_t)B_ * QL_ * D_) return;
  int dfull = (int)(idx % D_);
  int bq    = (int)(idx / D_);
  int b = bq / QL_, q = bq % QL_;
  int h = dfull / HD_;
  size_t sbase = ((size_t)b * H_ + h) * QL_ + q;

  float ms[NSEG], mstar = -1.0e38f;
#pragma unroll
  for (int s = 0; s < NSEG; ++s) {
    ms[s] = mpart[(size_t)s * (B_ * H_ * QL_) + sbase];
    mstar = fmaxf(mstar, ms[s]);
  }
  float l = 0.0f, o = 0.0f;
#pragma unroll
  for (int s = 0; s < NSEG; ++s) {
    float w = __expf(ms[s] - mstar);
    l += w * lpart[(size_t)s * (B_ * H_ * QL_) + sbase];
    o += w * Opart[(size_t)s * (B_ * QL_ * D_) + idx];
  }
  ao[idx] = (bf16_t)((l > 0.0f) ? o / l : 0.0f);
}

// ---------------------------------------------------------------------------
// Launch
// ---------------------------------------------------------------------------
extern "C" void kernel_launch(void* const* d_in, const int* in_sizes, int n_in,
                              void* d_out, int out_size, void* d_ws, size_t ws_size,
                              hipStream_t stream) {
  (void)in_sizes; (void)n_in; (void)out_size; (void)ws_size;

  const float* xq = (const float*)d_in[0];
  const float* xk = (const float*)d_in[1];
  const float* xv = (const float*)d_in[2];
  const unsigned char* mask = (const unsigned char*)d_in[3];
  const float* Wq = (const float*)d_in[4];
  const float* bq = (const float*)d_in[5];
  const float* Wk = (const float*)d_in[6];
  const float* bk = (const float*)d_in[7];
  const float* Wv = (const float*)d_in[8];
  const float* bv = (const float*)d_in[9];
  const float* Wf = (const float*)d_in[10];
  const float* bf = (const float*)d_in[11];

  const int NQ = B_ * QL_ * D_;          // 1,048,576
  const int NK = B_ * KL_ * D_;          // 16,777,216
  const int ND = D_ * D_;                // 1,048,576
  const int NW = B_ * QL_ * (KL_ / 64);  // mask words: 131,072
  const int NS = B_ * H_ * QL_;          // per-seg stat rows: 8192

  char* ws = (char*)d_ws;
  size_t off = 0;
  auto take = [&](size_t bytes) -> char* {
    off = (off + 255) & ~(size_t)255;
    char* p = ws + off;
    off += bytes;
    return p;
  };

  bf16_t* xq_b = (bf16_t*)take((size_t)NQ * 2);
  bf16_t* xk_b = (bf16_t*)take((size_t)NK * 2);
  bf16_t* xv_b = (bf16_t*)take((size_t)NK * 2);
  bf16_t* wq_b = (bf16_t*)take((size_t)ND * 2);
  bf16_t* wk_b = (bf16_t*)take((size_t)ND * 2);
  bf16_t* wv_b = (bf16_t*)take((size_t)ND * 2);
  bf16_t* wf_b = (bf16_t*)take((size_t)ND * 2);
  bf16_t* qp   = (bf16_t*)take((size_t)NQ * 2);
  bf16_t* kp   = (bf16_t*)take((size_t)NK * 2);
  bf16_t* vp   = (bf16_t*)take((size_t)NK * 2);
  bf16_t* ao   = (bf16_t*)take((size_t)NQ * 2);
  unsigned long long* mwords = (unsigned long long*)take((size_t)NW * 8);
  float* Opart = (float*)take((size_t)NSEG * NQ * 4);
  float* mpart = (float*)take((size_t)NSEG * NS * 4);
  float* lpart = (float*)take((size_t)NSEG * NS * 4);

  auto cvt = [&](const float* s, bf16_t* d, int n) {
    int n8 = n / 8;
    cvt_f32_bf16_kernel<<<(n8 + 255) / 256, 256, 0, stream>>>(s, d, n8);
  };
  cvt(xq, xq_b, NQ);
  cvt(xk, xk_b, NK);
  cvt(xv, xv_b, NK);
  cvt(Wq, wq_b, ND);
  cvt(Wk, wk_b, ND);
  cvt(Wv, wv_b, ND);
  cvt(Wf, wf_b, ND);
  pack_mask_kernel<<<(NW + 255) / 256, 256, 0, stream>>>(mask, mwords, NW);

  // Projections -> bf16 Q/K/V
  gemm_bias_kernel<false><<<dim3(D_ / 128, (B_ * QL_) / 128), 256, 0, stream>>>(
      xq_b, wq_b, bq, qp, B_ * QL_, D_, D_);
  gemm_bias_kernel<false><<<dim3(D_ / 128, (B_ * KL_) / 128), 256, 0, stream>>>(
      xk_b, wk_b, bk, kp, B_ * KL_, D_, D_);
  gemm_bias_kernel<false><<<dim3(D_ / 128, (B_ * KL_) / 128), 256, 0, stream>>>(
      xv_b, wv_b, bv, vp, B_ * KL_, D_, D_);

  // Split-K flash attention (512 workgroups) + combine
  flash_seg_kernel<<<NSEG * B_ * H_ * (QL_ / 128), 256, 0, stream>>>(
      qp, kp, vp, mwords, Opart, mpart, lpart);
  combine_kernel<<<(NQ + 255) / 256, 256, 0, stream>>>(Opart, mpart, lpart, ao);

  // Output projection -> fp32 result
  gemm_bias_kernel<true><<<dim3(D_ / 128, (B_ * QL_) / 128), 256, 0, stream>>>(
      ao, wf_b, bf, (float*)d_out, B_ * QL_, D_, D_);
}
